// TinyMiniLLMCLIPVLM_40029095199165
// MI455X (gfx1250) — compile-verified, matrix operand order checked by reading, not run on hardware
//
#include <hip/hip_runtime.h>
#include <hip/hip_bf16.h>
#include <stdint.h>

// ---------------- problem constants ----------------
#define PLACEHOLDER 31999
#define NUM_PH 128
constexpr int Bb = 8, SVv = 256, HVv = 1024, STt = 1024, Dd = 1024, Vv = 32000;
constexpr int KT_PROJ = HVv / 32;   // 32 k-tiles for projection
constexpr int KT_LM   = Dd  / 32;   // 32 k-tiles for LM head
constexpr int NT_PROJ = Dd  / 16;   // 64
constexpr int NT_LM   = Vv  / 16;   // 2000
constexpr int ROWS_LM = Bb * STt;   // 8192
constexpr int ROWS_VP = Bb * SVv;   // 2048

typedef __attribute__((ext_vector_type(16))) __bf16 v16bf;
typedef __attribute__((ext_vector_type(8)))  float  v8f;

struct alignas(16) U4 { uint32_t x, y, z, w; };
union Frag { U4 u[2]; v16bf v; };

__device__ __forceinline__ uint32_t bf16bits(float f) {
    uint32_t u = __float_as_uint(f);
    return (u + 0x7FFFu + ((u >> 16) & 1u)) >> 16;   // round-to-nearest-even
}
__device__ __forceinline__ uint32_t packbf(float lo, float hi) {
    return bf16bits(lo) | (bf16bits(hi) << 16);
}

// ---------------- K0: pack f32 (K x N, row-major) into WMMA-B bf16 fragments ----
// Fragment layout mirrors ISA 16-bit 16x32 operand layout:
//   lane L: n = nt*16 + L%16 ; half h = L/16
//   vgpr v<4 : k = kt*32 + h*8 + 2v   ; vgpr v>=4 : k = kt*32 + 16 + h*8 + 2(v-4)
// stored contiguously: out[((nt*KT + kt)*32 + lane)*8 + v]
__global__ void pack_b_kernel(const float* __restrict__ W, uint32_t* __restrict__ out,
                              int N, int ktiles, int total) {
    int t = blockIdx.x * blockDim.x + threadIdx.x;
    if (t >= total) return;
    int lane = t & 31;
    int kt = (t >> 5) % ktiles;
    int nt = (t >> 5) / ktiles;
    int n  = nt * 16 + (lane & 15);
    int kh = lane >> 4;
    uint32_t* o = out + (size_t)t * 8;
#pragma unroll
    for (int v = 0; v < 8; ++v) {
        int k = kt * 32 + ((v < 4) ? (kh * 8 + v * 2) : (16 + kh * 8 + (v - 4) * 2));
        float f0 = W[(size_t)k * N + n];
        float f1 = W[(size_t)(k + 1) * N + n];
        o[v] = packbf(f0, f1);
    }
}

// ---------------- K1: layernorm rows of vis_feats -> bf16 row-major -----------
__global__ void ln_kernel(const float* __restrict__ x, const float* __restrict__ g,
                          const float* __restrict__ be, uint16_t* __restrict__ out) {
    int row = blockIdx.x, tid = threadIdx.x;
    const float* xr = x + (size_t)row * HVv;
    __shared__ float red[256];
    float s = 0.f;
    for (int d = tid; d < HVv; d += 256) s += xr[d];
    red[tid] = s; __syncthreads();
    for (int off = 128; off > 0; off >>= 1) { if (tid < off) red[tid] += red[tid + off]; __syncthreads(); }
    float mu = red[0] * (1.0f / HVv); __syncthreads();
    float v = 0.f;
    for (int d = tid; d < HVv; d += 256) { float t = xr[d] - mu; v += t * t; }
    red[tid] = v; __syncthreads();
    for (int off = 128; off > 0; off >>= 1) { if (tid < off) red[tid] += red[tid + off]; __syncthreads(); }
    float rs = rsqrtf(red[0] * (1.0f / HVv) + 1e-5f);
    for (int d = tid; d < HVv; d += 256) {
        float y = (xr[d] - mu) * rs * g[d] + be[d];
        out[(size_t)row * HVv + d] = (uint16_t)bf16bits(y);
    }
}

// ---------------- K2: vis_proj = LN(vis) @ W_proj + b_proj (WMMA bf16) --------
__global__ void proj_kernel(const uint16_t* __restrict__ Abf, const uint32_t* __restrict__ Bp,
                            const float* __restrict__ bias, float* __restrict__ C) {
    int nt = blockIdx.x;          // 0..63
    int mt = blockIdx.y;          // 0..127
    int lane = threadIdx.x;       // 32 threads = 1 wave
    int nlo = lane & 15, h = lane >> 4;
    const uint16_t* ap = Abf + (size_t)(mt * 16 + nlo) * HVv + h * 8;
    const uint32_t* bp = Bp + (size_t)nt * KT_PROJ * 32 * 8;
    v8f c = {};
    for (int kt = 0; kt < KT_PROJ; ++kt) {
        Frag A, Bf;
        A.u[0] = *(const U4*)(ap + kt * 32);
        A.u[1] = *(const U4*)(ap + kt * 32 + 16);
        const U4* bu = (const U4*)(bp + ((size_t)kt * 32 + lane) * 8);
        Bf.u[0] = bu[0]; Bf.u[1] = bu[1];
        c = __builtin_amdgcn_wmma_f32_16x16x32_bf16(false, A.v, false, Bf.v, (short)0, c, false, false);
    }
    int n = nt * 16 + nlo;
    float bv = bias[n];
#pragma unroll
    for (int j = 0; j < 8; ++j) {
        int m = mt * 16 + j + 8 * h;
        C[(size_t)m * Dd + n] = c[j] + bv;
    }
}

// ---------------- K3: build text_emb (f32 + bf16) with placeholder splice -----
// SV=256, NUM_PH=128 -> each chunk is the mean of 2 consecutive vis_proj rows.
__global__ void temb_kernel(const int* __restrict__ ids, const float* __restrict__ vproj,
                            const float* __restrict__ etab, float* __restrict__ tef,
                            uint16_t* __restrict__ tebf) {
    int b = blockIdx.x >> 10;
    int s = blockIdx.x & 1023;
    __shared__ int sh_rank, sh_ph;
    if (threadIdx.x == 0) {
        const int* row = ids + b * STt;
        int cnt = 0;
        for (int t = 0; t <= s; ++t) cnt += (row[t] == PLACEHOLDER);
        sh_ph = (row[s] == PLACEHOLDER);
        sh_rank = cnt - 1;
    }
    __syncthreads();
    size_t obase = ((size_t)b * STt + s) * Dd;
    if (sh_ph && sh_rank >= 0 && sh_rank < NUM_PH) {
        const float* p0 = vproj + ((size_t)b * SVv + 2 * sh_rank) * Dd;
        const float* p1 = p0 + Dd;
        for (int d = threadIdx.x; d < Dd; d += 256) {
            float v = 0.5f * (p0[d] + p1[d]);
            tef[obase + d] = v;
            tebf[obase + d] = (uint16_t)bf16bits(v);
        }
    } else {
        const float* e = etab + (size_t)ids[b * STt + s] * Dd;
        for (int d = threadIdx.x; d < Dd; d += 256) {
            float v = e[d];
            tef[obase + d] = v;
            tebf[obase + d] = (uint16_t)bf16bits(v);
        }
    }
}

// ---------------- K4: LM head WMMA + online softmax + label-logit -------------
// Block = 8 waves over 64 rows: wave w -> rowtile rt=w%4, column phase p=w/2/2.
__global__ void lmhead_kernel(const uint16_t* __restrict__ Abf, const uint32_t* __restrict__ Bp,
                              const int* __restrict__ labels, float* __restrict__ rowLL) {
    int tid = threadIdx.x;
    int lane = tid & 31;
    int w = tid >> 5;
    int rt = w & 3;
    int p  = w >> 2;
    int rowBase = blockIdx.x * 64 + rt * 16;
    int nlo = lane & 15, h = lane >> 4;

    const uint16_t* ap = Abf + (size_t)(rowBase + nlo) * Dd + h * 8;

    int tgt[8];
#pragma unroll
    for (int j = 0; j < 8; ++j) {
        int row = rowBase + j + 8 * h;
        int s = row & 1023, b = row >> 10;
        tgt[j] = (s < 1023) ? labels[(b << 10) + s + 1] : -1;
    }

    float M[8], S[8], L[8];
#pragma unroll
    for (int j = 0; j < 8; ++j) { M[j] = -3.0e38f; S[j] = 0.f; L[j] = 0.f; }

    for (int nt = p; nt < NT_LM; nt += 2) {
        v8f c = {};
        const uint32_t* bp = Bp + (size_t)nt * KT_LM * 32 * 8;
#pragma unroll 4
        for (int kt = 0; kt < KT_LM; ++kt) {
            Frag A, Bf;
            A.u[0] = *(const U4*)(ap + kt * 32);
            A.u[1] = *(const U4*)(ap + kt * 32 + 16);
            const U4* bu = (const U4*)(bp + ((size_t)kt * 32 + lane) * 8);
            Bf.u[0] = bu[0]; Bf.u[1] = bu[1];
            c = __builtin_amdgcn_wmma_f32_16x16x32_bf16(false, A.v, false, Bf.v, (short)0, c, false, false);
        }
        int nG = nt * 16 + nlo;
#pragma unroll
        for (int j = 0; j < 8; ++j) {
            float v = c[j];
            float tm = v;                                   // row-max across 16 cols
            tm = fmaxf(tm, __shfl_xor(tm, 1));
            tm = fmaxf(tm, __shfl_xor(tm, 2));
            tm = fmaxf(tm, __shfl_xor(tm, 4));
            tm = fmaxf(tm, __shfl_xor(tm, 8));
            float nm = fmaxf(M[j], tm);
            float ts = __expf(v - nm);
            ts += __shfl_xor(ts, 1);
            ts += __shfl_xor(ts, 2);
            ts += __shfl_xor(ts, 4);
            ts += __shfl_xor(ts, 8);
            S[j] = S[j] * __expf(M[j] - nm) + ts;
            M[j] = nm;
            if (nG == tgt[j]) L[j] += v;
        }
    }
#pragma unroll
    for (int j = 0; j < 8; ++j) {
        float l = L[j];
        l += __shfl_xor(l, 1); l += __shfl_xor(l, 2);
        l += __shfl_xor(l, 4); l += __shfl_xor(l, 8);
        L[j] = l;
    }
    __shared__ float sM[2][64], sS[2][64], sL[2][64];
    if (nlo == 0) {
#pragma unroll
        for (int j = 0; j < 8; ++j) {
            int r = rt * 16 + 8 * h + j;
            sM[p][r] = M[j]; sS[p][r] = S[j]; sL[p][r] = L[j];
        }
    }
    __syncthreads();
    if (tid < 64) {
        int r = tid;
        float m0 = sM[0][r], m1 = sM[1][r];
        float m = fmaxf(m0, m1);
        float s = sS[0][r] * __expf(m0 - m) + sS[1][r] * __expf(m1 - m);
        float l = sL[0][r] + sL[1][r];
        int row = blockIdx.x * 64 + r;
        int spos = row & 1023;
        rowLL[row] = (spos < 1023) ? (l - (m + __logf(s))) : 0.f;
    }
}

// ---------------- K5a: per-batch normalized means -----------------------------
__global__ void means_kernel(const float* __restrict__ vproj, const float* __restrict__ tef,
                             const int* __restrict__ ids, const int* __restrict__ am,
                             float* __restrict__ vmean, float* __restrict__ tmean) {
    int b = blockIdx.x, tid = threadIdx.x;
    float vs[4] = {0, 0, 0, 0}, ts[4] = {0, 0, 0, 0};
    for (int s = 0; s < SVv; ++s) {
        const float* pr = vproj + ((size_t)b * SVv + s) * Dd;
#pragma unroll
        for (int q = 0; q < 4; ++q) vs[q] += pr[tid + 256 * q];
    }
    int cnt = 0;
    for (int s = 0; s < STt; ++s) {
        if (am[b * STt + s] != 0 && ids[b * STt + s] != PLACEHOLDER) {
            ++cnt;
            const float* pr = tef + ((size_t)b * STt + s) * Dd;
#pragma unroll
            for (int q = 0; q < 4; ++q) ts[q] += pr[tid + 256 * q];
        }
    }
    float denom = fmaxf((float)cnt, 1.0f);
    float ssv = 0.f, sst = 0.f;
#pragma unroll
    for (int q = 0; q < 4; ++q) {
        vs[q] *= (1.0f / SVv);
        ts[q] /= denom;
        ssv += vs[q] * vs[q];
        sst += ts[q] * ts[q];
    }
    __shared__ float red[256];
    red[tid] = ssv; __syncthreads();
    for (int off = 128; off > 0; off >>= 1) { if (tid < off) red[tid] += red[tid + off]; __syncthreads(); }
    float nv = fmaxf(sqrtf(red[0]), 1e-12f); __syncthreads();
    red[tid] = sst; __syncthreads();
    for (int off = 128; off > 0; off >>= 1) { if (tid < off) red[tid] += red[tid + off]; __syncthreads(); }
    float nt = fmaxf(sqrtf(red[0]), 1e-12f);
#pragma unroll
    for (int q = 0; q < 4; ++q) {
        vmean[(size_t)b * Dd + tid + 256 * q] = vs[q] / nv;
        tmean[(size_t)b * Dd + tid + 256 * q] = ts[q] / nt;
    }
}

// ---------------- K5b: losses -------------------------------------------------
__global__ void final_kernel(const float* __restrict__ rowLL, const float* __restrict__ vmean,
                             const float* __restrict__ tmean, float* __restrict__ out) {
    __shared__ float red[256];
    __shared__ float sim[64];
    int tid = threadIdx.x;
    float s = 0.f;
    for (int r = tid; r < ROWS_LM; r += 256) {
        if ((r & 1023) < 1023) s += rowLL[r];
    }
    red[tid] = s; __syncthreads();
    for (int off = 128; off > 0; off >>= 1) { if (tid < off) red[tid] += red[tid + off]; __syncthreads(); }
    float lmsum = red[0];
    __syncthreads();
    if (tid < 64) {
        int i = tid >> 3, j = tid & 7;
        const float* a = vmean + (size_t)i * Dd;
        const float* c = tmean + (size_t)j * Dd;
        float d = 0.f;
        for (int k = 0; k < Dd; ++k) d += a[k] * c[k];
        sim[tid] = d * (1.0f / 0.07f);
    }
    __syncthreads();
    if (tid == 0) {
        float lm = -lmsum / (8.0f * 1023.0f);
        float lr = 0.f, lc = 0.f;
        for (int i = 0; i < 8; ++i) {
            float mx = -3.0e38f;
            for (int j = 0; j < 8; ++j) mx = fmaxf(mx, sim[i * 8 + j]);
            float se = 0.f;
            for (int j = 0; j < 8; ++j) se += __expf(sim[i * 8 + j] - mx);
            lr += sim[i * 8 + i] - (mx + __logf(se));
        }
        for (int j = 0; j < 8; ++j) {
            float mx = -3.0e38f;
            for (int i = 0; i < 8; ++i) mx = fmaxf(mx, sim[i * 8 + j]);
            float se = 0.f;
            for (int i = 0; i < 8; ++i) se += __expf(sim[i * 8 + j] - mx);
            lc += sim[j * 8 + j] - (mx + __logf(se));
        }
        float cont = -(lr + lc) / 16.0f;
        float total = lm + 0.5f * cont;
        out[0] = total; out[1] = lm; out[2] = cont;
    }
}

// ---------------- launcher ----------------------------------------------------
extern "C" void kernel_launch(void* const* d_in, const int* in_sizes, int n_in,
                              void* d_out, int out_size, void* d_ws, size_t ws_size,
                              hipStream_t stream) {
    const float* vis_feats = (const float*)d_in[0];
    const int*   input_ids = (const int*)d_in[1];
    const int*   attn      = (const int*)d_in[2];
    const int*   labels    = (const int*)d_in[3];
    const float* ln_gamma  = (const float*)d_in[4];
    const float* ln_beta   = (const float*)d_in[5];
    const float* W_proj    = (const float*)d_in[6];
    const float* b_proj    = (const float*)d_in[7];
    const float* etab      = (const float*)d_in[8];
    const float* W_lm      = (const float*)d_in[9];
    float* out = (float*)d_out;

    char* ws = (char*)d_ws;
    size_t off = 0;
    auto alloc = [&](size_t bytes) -> void* {
        void* p = ws + off;
        off += (bytes + 255) & ~(size_t)255;
        return p;
    };
    uint32_t* wlm_p = (uint32_t*)alloc((size_t)NT_LM * KT_LM * 32 * 8 * 4);     // 62.5 MB
    uint32_t* wpr_p = (uint32_t*)alloc((size_t)NT_PROJ * KT_PROJ * 32 * 8 * 4); // 2 MB
    uint16_t* vn_bf = (uint16_t*)alloc((size_t)ROWS_VP * HVv * 2);              // 4 MB
    float*    vproj = (float*)alloc((size_t)ROWS_VP * Dd * 4);                  // 8 MB
    float*    tef   = (float*)alloc((size_t)ROWS_LM * Dd * 4);                  // 32 MB
    uint16_t* tebf  = (uint16_t*)alloc((size_t)ROWS_LM * Dd * 2);               // 16 MB
    float*    rowLL = (float*)alloc((size_t)ROWS_LM * 4);
    float*    vmean = (float*)alloc((size_t)Bb * Dd * 4);
    float*    tmean = (float*)alloc((size_t)Bb * Dd * 4);

    int totLM = NT_LM * KT_LM * 32;
    pack_b_kernel<<<(totLM + 255) / 256, 256, 0, stream>>>(W_lm, wlm_p, Vv, KT_LM, totLM);
    int totPR = NT_PROJ * KT_PROJ * 32;
    pack_b_kernel<<<(totPR + 255) / 256, 256, 0, stream>>>(W_proj, wpr_p, Dd, KT_PROJ, totPR);

    ln_kernel<<<ROWS_VP, 256, 0, stream>>>(vis_feats, ln_gamma, ln_beta, vn_bf);
    proj_kernel<<<dim3(NT_PROJ, ROWS_VP / 16), 32, 0, stream>>>(vn_bf, wpr_p, b_proj, vproj);
    temb_kernel<<<Bb * STt, 256, 0, stream>>>(input_ids, vproj, etab, tef, tebf);
    lmhead_kernel<<<ROWS_LM / 64, 256, 0, stream>>>(tebf, wlm_p, labels, rowLL);
    means_kernel<<<Bb, 256, 0, stream>>>(vproj, tef, input_ids, attn, vmean, tmean);
    final_kernel<<<1, 256, 0, stream>>>(rowLL, vmean, tmean, out);
}
